// DebertaV3CRFForTokenClassification_47287589929089
// MI455X (gfx1250) — matrix operand order repaired and checked by reading, not compiled, not run on hardware
//
#include <hip/hip_runtime.h>
#include <hip/hip_bf16.h>
#include <math.h>

// ---------------------------------------------------------------------------
// Types for WMMA operands (gfx1250, wave32)
// ---------------------------------------------------------------------------
typedef __attribute__((ext_vector_type(16))) __bf16 bf16x16;
typedef __attribute__((ext_vector_type(8)))  float  f32x8;
typedef __attribute__((ext_vector_type(4)))  int    i32x4;

union BF16Frag {
    i32x4   q[2];   // two 16-byte loads
    bf16x16 v;      // WMMA operand
};

// round-to-nearest-even f32 -> bf16 (bit-level, no __bf16 arithmetic needed)
__device__ __forceinline__ unsigned short f2bf(float f) {
    union { float f; unsigned int u; } v; v.f = f;
    unsigned int u = v.u;
    unsigned int r = u + 0x7FFFu + ((u >> 16) & 1u);
    return (unsigned short)(r >> 16);
}

#define Bb 64
#define Ss 512
#define Hh 1024
#define Ll 9

// ---------------------------------------------------------------------------
// Prep: W1 [1024][512] f32 -> W1T [512][1024] bf16 (transposed, row = out col)
// ---------------------------------------------------------------------------
__global__ void prep_w1t(const float* __restrict__ W1, unsigned short* __restrict__ w1t) {
    int gid = blockIdx.x * blockDim.x + threadIdx.x;
    if (gid < 512 * 1024) {
        int n = gid >> 10;       // output column 0..511
        int k = gid & 1023;      // input dim 0..1023
        w1t[gid] = f2bf(W1[(size_t)k * 512 + n]);
    }
}

// Prep: W2 [512][9] f32 -> W2T [16][512] bf16, rows 9..15 zero-padded
__global__ void prep_w2t(const float* __restrict__ W2, unsigned short* __restrict__ w2t) {
    int gid = blockIdx.x * blockDim.x + threadIdx.x;
    if (gid < 16 * 512) {
        int n = gid >> 9;        // padded label 0..15
        int k = gid & 511;       // hidden dim 0..511
        w2t[gid] = (n < Ll) ? f2bf(W2[(size_t)k * Ll + n]) : (unsigned short)0;
    }
}

// ---------------------------------------------------------------------------
// Fused: LayerNorm -> GEMM1(bf16 WMMA)+GELU -> GEMM2(bf16 WMMA) -> emissions
// Block: 128 threads (4 waves), 16 tokens per block.
// emissions layout: em[s][b][l]  (time-major for the CRF kernel)
// ---------------------------------------------------------------------------
__global__ void __launch_bounds__(128)
fused_ln_mlp_emissions(const float* __restrict__ hidden,
                       const float* __restrict__ gamma,
                       const float* __restrict__ beta,
                       const unsigned short* __restrict__ w1t,
                       const float* __restrict__ b1,
                       const unsigned short* __restrict__ w2t,
                       const float* __restrict__ b2,
                       float* __restrict__ em) {
    __shared__ unsigned short xs[16][1024];   // normalized activations, bf16 (32 KB)
    __shared__ unsigned short hs[16][512];    // GELU(h), bf16 (16 KB)

    const int tid  = threadIdx.x;
    const int w    = tid >> 5;        // wave 0..3
    const int lane = tid & 31;
    const int n0   = lane & 15;       // column-within-tile / A row index
    const int hi   = lane >> 4;       // lane half (WMMA layout selector)
    const int tileTok = blockIdx.x * 16;

    // ---------------- Phase 1: LayerNorm (each wave: 4 tokens) ----------------
    for (int j = 0; j < 4; ++j) {
        const int t = w * 4 + j;                                   // local token
        const float* row = hidden + (size_t)(tileTok + t) * Hh;
        float v[32];
        float s = 0.f;
        #pragma unroll
        for (int i = 0; i < 32; ++i) { v[i] = row[i * 32 + lane]; s += v[i]; }
        #pragma unroll
        for (int off = 16; off > 0; off >>= 1) s += __shfl_xor(s, off);
        const float mu = s * (1.0f / 1024.0f);
        float q = 0.f;
        #pragma unroll
        for (int i = 0; i < 32; ++i) { float d = v[i] - mu; q += d * d; }
        #pragma unroll
        for (int off = 16; off > 0; off >>= 1) q += __shfl_xor(q, off);
        const float rstd = rsqrtf(q * (1.0f / 1024.0f) + 1e-5f);
        #pragma unroll
        for (int i = 0; i < 32; ++i) {
            const int idx = i * 32 + lane;
            const float y = (v[i] - mu) * rstd * gamma[idx] + beta[idx];
            xs[t][idx] = f2bf(y);
        }
    }
    __syncthreads();

    // ---------------- Phase 2: GEMM1 16x512 = xs(16x1024) @ W1T^T -------------
    // Each wave owns 128 output columns = 8 WMMA n-tiles; K loop: 32 steps of 32.
    const int cbase = w * 128;
    f32x8 acc[8] = {};

    for (int kk = 0; kk < 32; ++kk) {
        const int kb = kk * 32;
        // A fragment: 16-bit A layout — lane holds row n0, two contiguous 8-half runs
        BF16Frag a;
        a.q[0] = *(const i32x4*)&xs[n0][kb + hi * 8];
        a.q[1] = *(const i32x4*)&xs[n0][kb + 16 + hi * 8];
        // prefetch next K-slab of this wave's W1T panel
        if (kk + 1 < 32)
            __builtin_prefetch(w1t + (size_t)(cbase + n0) * 1024 + kb + 32, 0, 1);
        #pragma unroll
        for (int nt = 0; nt < 8; ++nt) {
            const unsigned short* bp =
                w1t + (size_t)(cbase + nt * 16 + n0) * 1024 + kb + hi * 8;
            BF16Frag bfr;
            bfr.q[0] = *(const i32x4*)(bp);
            bfr.q[1] = *(const i32x4*)(bp + 16);
            acc[nt] = __builtin_amdgcn_wmma_f32_16x16x32_bf16(
                false, a.v, false, bfr.v, (short)0, acc[nt], false, false);
        }
    }

    // bias + exact GELU, write bf16 h to LDS (C layout: vgpr r -> row r + 8*hi)
    #pragma unroll
    for (int nt = 0; nt < 8; ++nt) {
        const int col = cbase + nt * 16 + n0;
        const float bias = b1[col];
        #pragma unroll
        for (int r = 0; r < 8; ++r) {
            const int row = r + hi * 8;
            const float x = acc[nt][r] + bias;
            const float g = 0.5f * x * (1.0f + erff(x * 0.70710678118654752f));
            hs[row][col] = f2bf(g);
        }
    }
    __syncthreads();

    // ---------------- Phase 3: GEMM2 16x16 = hs(16x512) @ W2T^T (wave 0) ------
    if (w == 0) {
        f32x8 acc2 = {};
        for (int kk = 0; kk < 16; ++kk) {
            const int kb = kk * 32;
            BF16Frag a, bfr;
            a.q[0] = *(const i32x4*)&hs[n0][kb + hi * 8];
            a.q[1] = *(const i32x4*)&hs[n0][kb + 16 + hi * 8];
            const unsigned short* bp = w2t + (size_t)n0 * 512 + kb + hi * 8;
            bfr.q[0] = *(const i32x4*)(bp);
            bfr.q[1] = *(const i32x4*)(bp + 16);
            acc2 = __builtin_amdgcn_wmma_f32_16x16x32_bf16(
                false, a.v, false, bfr.v, (short)0, acc2, false, false);
        }
        if (n0 < Ll) {
            const float bias = b2[n0];
            #pragma unroll
            for (int r = 0; r < 8; ++r) {
                const int tok = tileTok + r + hi * 8;
                const int b = tok >> 9;          // / S
                const int s = tok & 511;         // % S
                em[((size_t)s * Bb + b) * Ll + n0] = acc2[r] + bias;
            }
        }
    }
}

// ---------------------------------------------------------------------------
// CRF NLL: one thread per batch element; block reduce -> -mean(llh)
// ---------------------------------------------------------------------------
__global__ void __launch_bounds__(64)
crf_nll(const float* __restrict__ em,          // [S][B][L]
        const float* __restrict__ start_trans, // [L]
        const float* __restrict__ end_trans,   // [L]
        const float* __restrict__ trans,       // [L][L]
        const int* __restrict__ labels,        // [B][S]
        const int* __restrict__ attn,          // [B][S]
        float* __restrict__ out) {
    __shared__ float red[Bb];
    const int b = threadIdx.x;

    float tr[Ll * Ll], st[Ll], en[Ll];
    #pragma unroll
    for (int i = 0; i < Ll * Ll; ++i) tr[i] = trans[i];
    #pragma unroll
    for (int i = 0; i < Ll; ++i) { st[i] = start_trans[i]; en[i] = end_trans[i]; }

    const int* lab = labels + (size_t)b * Ss;
    const int* am  = attn   + (size_t)b * Ss;

    // t = 0 (mask[:,0] forced true)
    const int l0  = lab[0];
    const int tg0 = (l0 == -100) ? 0 : l0;
    float sc[Ll];
    #pragma unroll
    for (int j = 0; j < Ll; ++j) sc[j] = st[j] + em[((size_t)0 * Bb + b) * Ll + j];
    float num = st[tg0] + em[((size_t)0 * Bb + b) * Ll + tg0];
    int prev = tg0;
    int cnt  = 1;

    for (int s = 1; s < Ss; ++s) {
        const int ls = lab[s];
        const int tg = (ls == -100) ? 0 : ls;
        const bool mk = (ls != -100) && (am[s] == 1);
        float er[Ll];
        #pragma unroll
        for (int j = 0; j < Ll; ++j) er[j] = em[((size_t)s * Bb + b) * Ll + j];
        if (mk) { num += tr[prev * Ll + tg] + er[tg]; ++cnt; }
        prev = tg;
        // forward recursion: nx[j] = er[j] + LSE_i(sc[i] + tr[i][j])
        float nx[Ll];
        #pragma unroll
        for (int j = 0; j < Ll; ++j) {
            float mx = -3.4e38f;
            #pragma unroll
            for (int i = 0; i < Ll; ++i) {
                const float t2 = sc[i] + tr[i * Ll + j];
                mx = fmaxf(mx, t2);
            }
            float ssum = 0.f;
            #pragma unroll
            for (int i = 0; i < Ll; ++i)
                ssum += __expf(sc[i] + tr[i * Ll + j] - mx);
            nx[j] = mx + __logf(ssum) + er[j];
        }
        if (mk) {
            #pragma unroll
            for (int j = 0; j < Ll; ++j) sc[j] = nx[j];
        }
    }

    // end transition on tag at position (cnt-1), per reference semantics
    const int lpos = cnt - 1;
    int lt = lab[lpos]; lt = (lt == -100) ? 0 : lt;
    num += en[lt];

    float mx = -3.4e38f;
    #pragma unroll
    for (int j = 0; j < Ll; ++j) mx = fmaxf(mx, sc[j] + en[j]);
    float ssum = 0.f;
    #pragma unroll
    for (int j = 0; j < Ll; ++j) ssum += __expf(sc[j] + en[j] - mx);
    const float denom = mx + __logf(ssum);

    red[b] = num - denom;
    __syncthreads();
    if (b == 0) {
        float t = 0.f;
        for (int i = 0; i < Bb; ++i) t += red[i];
        out[0] = -t * (1.0f / Bb);
    }
}

// ---------------------------------------------------------------------------
// Launch
// ---------------------------------------------------------------------------
extern "C" void kernel_launch(void* const* d_in, const int* in_sizes, int n_in,
                              void* d_out, int out_size, void* d_ws, size_t ws_size,
                              hipStream_t stream) {
    const float* hidden = (const float*)d_in[0];
    const float* gamma  = (const float*)d_in[1];
    const float* beta   = (const float*)d_in[2];
    const float* W1     = (const float*)d_in[3];
    const float* b1     = (const float*)d_in[4];
    const float* W2     = (const float*)d_in[5];
    const float* b2     = (const float*)d_in[6];
    const float* stt    = (const float*)d_in[7];
    const float* ent    = (const float*)d_in[8];
    const float* trn    = (const float*)d_in[9];
    const int*   labels = (const int*)d_in[10];
    const int*   attn   = (const int*)d_in[11];
    float* out = (float*)d_out;

    // workspace layout
    unsigned char* ws = (unsigned char*)d_ws;
    unsigned short* w1t = (unsigned short*)(ws);                       // 512*1024*2 = 1 MB
    unsigned short* w2t = (unsigned short*)(ws + 512 * 1024 * 2);      // 16*512*2   = 16 KB
    float*          em  = (float*)(ws + 512 * 1024 * 2 + 16 * 512 * 2); // S*B*L*4 ≈ 1.13 MB

    prep_w1t<<<(512 * 1024 + 255) / 256, 256, 0, stream>>>(W1, w1t);
    prep_w2t<<<(16 * 512 + 255) / 256, 256, 0, stream>>>(W2, w2t);

    const int nTiles = (Bb * Ss) / 16;   // 2048
    fused_ln_mlp_emissions<<<nTiles, 128, 0, stream>>>(
        hidden, gamma, beta, w1t, b1, w2t, b2, em);

    crf_nll<<<1, Bb, 0, stream>>>(em, stt, ent, trn, labels, attn, out);
}
